// AttZAM_32091995636391
// MI455X (gfx1250) — compile-verified, matrix operand order checked by reading, not run on hardware
//
#include <hip/hip_runtime.h>
#include <math.h>

// Problem constants (from the reference): BSZ=256, MAX_LEN=1024, D=256, D_ATTN=128
#define BSZ     256
#define MAXLEN  1024
#define DDIM    256
#define DATTN   128

typedef __attribute__((ext_vector_type(16))) __bf16 v16bf;
typedef __attribute__((ext_vector_type(8)))  float  v8f;

// Split an fp32 value into bf16 hi + bf16 lo such that hi+lo ~= x.
// Products hi*hi + hi*lo + lo*hi recover ~fp32-accurate dot products
// while using the fast V_WMMA_F32_16X16X32_BF16 path.
__device__ __forceinline__ void load_split16(const float* __restrict__ p, int half,
                                             v16bf& hi, v16bf& lo) {
    // 16-bit A/B fragment layout (CDNA5 ISA 7.12.2): per lane, elements 0..7
    // are K = 8*half + i, elements 8..15 are K = 16 + 8*half + (i-8).
    const float4 f0 = *(const float4*)(p + 8 * half);
    const float4 f1 = *(const float4*)(p + 8 * half + 4);
    const float4 f2 = *(const float4*)(p + 16 + 8 * half);
    const float4 f3 = *(const float4*)(p + 16 + 8 * half + 4);
    float t[16];
    t[0]=f0.x;  t[1]=f0.y;  t[2]=f0.z;  t[3]=f0.w;
    t[4]=f1.x;  t[5]=f1.y;  t[6]=f1.z;  t[7]=f1.w;
    t[8]=f2.x;  t[9]=f2.y;  t[10]=f2.z; t[11]=f2.w;
    t[12]=f3.x; t[13]=f3.y; t[14]=f3.z; t[15]=f3.w;
#pragma unroll
    for (int i = 0; i < 16; ++i) {
        __bf16 h = (__bf16)t[i];
        hi[i] = h;
        lo[i] = (__bf16)(t[i] - (float)h);
    }
}

// ---------------------------------------------------------------------------
// Phase A: fused  v[b,j] = sum_a tanh( (emb_q[b,:] . w_f[j,a,:]) + b_f[j,a] ) * w_h[a]
// grid = (16 b-tiles, 256 j), block = 256 threads = 8 waves.
// Wave w computes the 16x16 WMMA tile (b-tile x a-tile=w) with a bf16-split
// K=256 GEMM, applies bias/tanh/w_h, reduces over 'a' and combines via LDS.
// ---------------------------------------------------------------------------
__global__ __launch_bounds__(256) void attn_phaseA(
    const float* __restrict__ emb_q,   // [256, 256]
    const float* __restrict__ w_f,     // [256, 128, 256] (j, a, k), k contiguous
    const float* __restrict__ b_f,     // [256, 128]
    const float* __restrict__ w_h,     // [128]
    float* __restrict__ v)             // [256, 256] (b, j)
{
    const int tid  = threadIdx.x;
    const int wave = tid >> 5;
    const int lane = tid & 31;
    const int m16  = lane & 15;   // row within tile (A) / col within tile (B)
    const int half = lane >> 4;   // K-half selector per fragment layout

    const int b0 = blockIdx.x * 16;   // batch tile base
    const int j  = blockIdx.y;        // output feature j
    const int a0 = wave * 16;         // this wave's a-tile base

    const float* Arow = emb_q + (size_t)(b0 + m16) * DDIM;                    // A: M=b, K=k
    const float* Brow = w_f + ((size_t)j * DATTN + (size_t)(a0 + m16)) * DDIM; // B: N=a, K=k

    v8f acc = {};
#pragma unroll
    for (int kk = 0; kk < DDIM / 32; ++kk) {
        const int k0 = kk * 32;
        v16bf ahi, alo, bhi, blo;
        load_split16(Arow + k0, half, ahi, alo);
        load_split16(Brow + k0, half, bhi, blo);
        // hi*hi + hi*lo + lo*hi  ->  ~fp32 accuracy on the bf16 WMMA path
        acc = __builtin_amdgcn_wmma_f32_16x16x32_bf16(false, ahi, false, bhi,
                                                      (short)0, acc, false, false);
        acc = __builtin_amdgcn_wmma_f32_16x16x32_bf16(false, ahi, false, blo,
                                                      (short)0, acc, false, false);
        acc = __builtin_amdgcn_wmma_f32_16x16x32_bf16(false, alo, false, bhi,
                                                      (short)0, acc, false, false);
    }

    // C/D layout: VGPR r, lanes 0-15 -> (M=r, N=lane); lanes 16-31 -> (M=r+8, N=lane-16)
    const int   a_idx = a0 + m16;                 // N maps to the 'a' index
    const float bias  = b_f[(size_t)j * DATTN + a_idx];
    const float wh    = w_h[a_idx];

    __shared__ float red[8 * 16];                 // [wave][M]
#pragma unroll
    for (int r = 0; r < 8; ++r) {
        float t = tanhf(acc[r] + bias) * wh;
        // reduce over N (the 'a' direction): masks < 16 stay within each half
        t += __shfl_xor(t, 1, 32);
        t += __shfl_xor(t, 2, 32);
        t += __shfl_xor(t, 4, 32);
        t += __shfl_xor(t, 8, 32);
        if (m16 == 0) red[wave * 16 + half * 8 + r] = t;  // M = r + 8*half
    }
    __syncthreads();

    if (tid < 16) {
        float s = 0.f;
#pragma unroll
        for (int w = 0; w < 8; ++w) s += red[w * 16 + tid];
        v[(size_t)(b0 + tid) * DDIM + j] = s;
    }
}

// ---------------------------------------------------------------------------
// Phase B: weight[b,l] = sum_j emb_iseq[b,l,j] * v[b,j]
// Pure bandwidth: 256 MB of emb_iseq @ 23.3 TB/s ~ 11 us floor.
// Wave-cooperative rows: 32 lanes cover j (2 float4s each) -> fully coalesced
// 1 KB per row per wave; v[b,*] lives in registers (8 floats/lane).
// grid = (4 l-chunks, 256 b), block = 256 threads = 8 waves, 32 rows/wave.
// ---------------------------------------------------------------------------
__global__ __launch_bounds__(256) void attn_phaseB(
    const float* __restrict__ emb_iseq, // [256, 1024, 256]
    const float* __restrict__ v,        // [256, 256]
    float* __restrict__ out)            // [256, 1024]
{
    const int tid  = threadIdx.x;
    const int wave = tid >> 5;
    const int lane = tid & 31;
    const int b    = blockIdx.y;
    const int l0   = blockIdx.x * 256 + wave * 32;

    // each lane owns j = lane*8 .. lane*8+7 of v[b,:]
    const float4* vv = (const float4*)(v + (size_t)b * DDIM);
    const float4  w0 = vv[lane * 2];
    const float4  w1 = vv[lane * 2 + 1];

    const float4* ebase = (const float4*)(emb_iseq + ((size_t)b * MAXLEN + l0) * DDIM);
    float* obase = out + (size_t)b * MAXLEN + l0;

#pragma unroll 4
    for (int i = 0; i < 32; ++i) {
        const float4* row = ebase + (size_t)i * (DDIM / 4);
        const float4 e0 = row[lane * 2];
        const float4 e1 = row[lane * 2 + 1];
        float s = e0.x * w0.x + e0.y * w0.y + e0.z * w0.z + e0.w * w0.w
                + e1.x * w1.x + e1.y * w1.y + e1.z * w1.z + e1.w * w1.w;
        s += __shfl_xor(s, 1, 32);
        s += __shfl_xor(s, 2, 32);
        s += __shfl_xor(s, 4, 32);
        s += __shfl_xor(s, 8, 32);
        s += __shfl_xor(s, 16, 32);
        if (lane == 0) obase[i] = s;
    }
}

extern "C" void kernel_launch(void* const* d_in, const int* in_sizes, int n_in,
                              void* d_out, int out_size, void* d_ws, size_t ws_size,
                              hipStream_t stream) {
    const float* emb_q    = (const float*)d_in[0]; // [256,256]
    const float* emb_iseq = (const float*)d_in[1]; // [256,1024,256]
    const float* w_f      = (const float*)d_in[2]; // [256,128,256]
    const float* b_f      = (const float*)d_in[3]; // [256,128]
    const float* w_h      = (const float*)d_in[4]; // [128] (w_h[:,0])
    float* out = (float*)d_out;                    // [256,1024]
    float* v   = (float*)d_ws;                     // [256,256] scratch (256 KB)

    dim3 gA(BSZ / 16, DDIM);        // (16 b-tiles, 256 j)
    attn_phaseA<<<gA, 256, 0, stream>>>(emb_q, w_f, b_f, w_h, v);

    dim3 gB(MAXLEN / 256, BSZ);     // (4 l-chunks, 256 b)
    attn_phaseB<<<gB, 256, 0, stream>>>(emb_iseq, v, out);
}
